// Athena_24739011625811
// MI455X (gfx1250) — matrix operand order repaired and checked by reading, not compiled, not run on hardware
//
#include <hip/hip_runtime.h>
#include <hip/hip_bf16.h>
#include <stdint.h>

// ---- model dims ----
#define S_    2048
#define E_    2048
#define HID_  8192
#define L_    4
#define H_    16
#define KV_   4
#define DK_   128
#define DH_   128
#define WIN_  1024
#define QDIM  (H_ * DK_)   // 2048
#define KDIM  (KV_ * DK_)  // 512
#define VDIM  (KV_ * DH_)  // 512
#define ODIM  (H_ * DH_)   // 2048
#define VOC_  32000

typedef __attribute__((ext_vector_type(16))) __bf16 v16bf;
typedef __attribute__((ext_vector_type(8)))  float  v8f;
typedef unsigned int u32;
typedef __attribute__((ext_vector_type(4))) u32 v4u;
typedef __attribute__((ext_vector_type(8))) int v8i;
typedef __attribute__((ext_vector_type(4))) int v4i;

union Frag { uint32_t u[8]; v16bf v; };

#if defined(__has_builtin)
#  if __has_builtin(__builtin_amdgcn_tensor_load_to_lds) && \
      __has_builtin(__builtin_amdgcn_s_wait_tensorcnt)
#    define USE_TDM 1
#  endif
#endif
#ifndef USE_TDM
#  define USE_TDM 0
#endif

// A-operand (and P) 16-bit fragment K-offsets (ISA 7.12.2, 16-bit A 16x32):
__device__ __forceinline__ int aoff(int j, int kh) {
  return (j < 4) ? (kh * 8 + 2 * j) : (16 + kh * 8 + 2 * (j - 4));
}
// B-operand 16-bit fragment: same K order as A (lanes 0-15 K=0..7,16..23; lanes 16-31 K=8..15,24..31)
__device__ __forceinline__ int boff(int j, int kh) {
  return (j < 4) ? (kh * 8 + 2 * j) : (16 + kh * 8 + 2 * (j - 4));
}

#if USE_TDM
// Tensor Data Mover: load a [tile_rows x tile_cols] bf16 tile (row stride
// row_stride_elems) from global into LDS at lds_byte_addr. D# packed per
// CDNA5 ISA 8.3/8.4 (group0: count/lds/global/type, group1: data_size=2B,
// dims, tile dims, dim0 stride). Issued by one wave; EXEC is ignored.
// This toolchain exposes the 6-arg builtin (g0, g1, g2, g3, g_extra, cpol).
__device__ __forceinline__ void tdm_load_2d_bf16(
    const __bf16* gsrc, u32 lds_byte_addr, u32 tile_cols, u32 tile_rows,
    u32 row_stride_elems) {
  uint64_t ga = (uint64_t)(uintptr_t)gsrc;
  v4u g0;
  g0.x = 1u;                                            // count=1 (valid)
  g0.y = lds_byte_addr;                                 // lds_addr
  g0.z = (u32)ga;                                       // global_addr[31:0]
  g0.w = (u32)((ga >> 32) & 0x01FFFFFFu) | (2u << 30);  // [56:32] | type=2
  u32 dim0 = row_stride_elems;                          // tensor_dim0
  u32 dim1 = 1u << 20;                                  // tensor_dim1 (OOB only)
  v8i g1;
  g1[0] = (int)(1u << 16);                              // data_size=1 -> 2B
  g1[1] = (int)((dim0 & 0xFFFFu) << 16);                // dim0 lo16
  g1[2] = (int)((dim0 >> 16) | ((dim1 & 0xFFFFu) << 16));
  g1[3] = (int)((dim1 >> 16) | (tile_cols << 16));      // tile_dim0
  g1[4] = (int)(tile_rows & 0xFFFFu);                   // tile_dim1
  g1[5] = (int)row_stride_elems;                        // dim0_stride lo32
  g1[6] = 0;
  g1[7] = 0;
  v4i zz4 = {0, 0, 0, 0};
  v8i zz8 = {0, 0, 0, 0, 0, 0, 0, 0};
  __builtin_amdgcn_tensor_load_to_lds(g0, g1, zz4, zz4, zz8, 0);
}
#endif

// ------------------------------------------------------------------
__global__ __launch_bounds__(256) void embed_kernel(
    const int* __restrict__ tokens, const float* __restrict__ table,
    float* __restrict__ x) {
  int s = blockIdx.x;
  int tok = tokens[s];
  float* xr = x + (size_t)s * E_;
  const float* tr = table + (size_t)tok * E_;
  for (int e = threadIdx.x; e < E_; e += 256)
    xr[e] = (tok == 0) ? 0.0f : tr[e];
}

// ------------------------------------------------------------------
__global__ __launch_bounds__(256) void rmsnorm_kernel(
    const float* __restrict__ x, __bf16* __restrict__ n) {
  __shared__ float red[256];
  int s = blockIdx.x, tid = threadIdx.x;
  const float* row = x + (size_t)s * E_;
  float acc = 0.0f;
  for (int e = tid; e < E_; e += 256) { float v = row[e]; acc += v * v; }
  red[tid] = acc;
  __syncthreads();
  for (int st = 128; st > 0; st >>= 1) {
    if (tid < st) red[tid] += red[tid + st];
    __syncthreads();
  }
  float scale = rsqrtf(red[0] * (1.0f / (float)E_) + 1e-5f);
  __bf16* orow = n + (size_t)s * E_;
  for (int e = tid; e < E_; e += 256) orow[e] = (__bf16)(row[e] * scale);
}

// ------------------------------------------------------------------
// GEMM: C[M,N] (+)= A_bf16[M,K] * B_f32[K,N].
// Block tile 128x128x32, 8 waves (2x4). A tiles via TDM (double-buffered
// LDS, tensorcnt-pipelined); B fp32 weights register-pipelined and
// converted to bf16 on the way into LDS; stream prefetched 2 tiles ahead.
// ------------------------------------------------------------------
template <int ADD>
__global__ __launch_bounds__(256) void gemm_kernel(
    const __bf16* __restrict__ A, const float* __restrict__ B,
    float* __restrict__ C, int K, int ldb, int ldc) {
  __shared__ __bf16 lds_a[2][128][32];  // [buf][m][k]
  __shared__ __bf16 lds_b[128][32];     // [n][k]
  const int tid  = threadIdx.x;
  const int lane = tid & 31;
  const int wave = tid >> 5;
  const int wm = wave >> 2;
  const int wn = wave & 3;
  const int lr = lane & 15;
  const int kh = lane >> 4;
  const int bm = blockIdx.y * 128;
  const int bn = blockIdx.x * 128;

  v8f zero8 = {0, 0, 0, 0, 0, 0, 0, 0};
  v8f acc[4][2];
#pragma unroll
  for (int mi = 0; mi < 4; ++mi)
#pragma unroll
    for (int ni = 0; ni < 2; ++ni) acc[mi][ni] = zero8;

  const int arow = tid >> 1;
  const int acol = (tid & 1) * 16;
  const int brow = tid >> 3;
  const int bcol = (tid & 7) * 16;

  const __bf16* atile = A + (size_t)bm * K;
  const float*  bbase = B + bn + bcol;

#if USE_TDM
  if (wave == 0)
    tdm_load_2d_bf16(atile, (u32)(uintptr_t)&lds_a[0][0][0], 32, 128, (u32)K);
#else
  uint4 aR = *(const uint4*)(atile + (size_t)arow * K + acol);
#endif
  float4 bR[4];
  {
    const float4* bp = (const float4*)(bbase + (size_t)brow * ldb);
#pragma unroll
    for (int t = 0; t < 4; ++t) bR[t] = bp[t];
  }

  const int niter = K >> 5;
  for (int i = 0;;) {
    const int buf = i & 1;
    // commit staged B tile (and A for the non-TDM path)
#pragma unroll
    for (int t = 0; t < 4; ++t) {
      lds_b[bcol + 4 * t + 0][brow] = (__bf16)bR[t].x;
      lds_b[bcol + 4 * t + 1][brow] = (__bf16)bR[t].y;
      lds_b[bcol + 4 * t + 2][brow] = (__bf16)bR[t].z;
      lds_b[bcol + 4 * t + 3][brow] = (__bf16)bR[t].w;
    }
#if USE_TDM
    if (wave == 0) __builtin_amdgcn_s_wait_tensorcnt(0);  // A tile i landed
#else
    *(uint4*)&lds_a[buf][arow][acol] = aR;
#endif
    __syncthreads();

    // stage tile i+1 while the WMMAs below execute
    if (i + 1 < niter) {
      const int k0 = (i + 1) << 5;
#if USE_TDM
      if (wave == 0)
        tdm_load_2d_bf16(atile + k0, (u32)(uintptr_t)&lds_a[buf ^ 1][0][0],
                         32, 128, (u32)K);
#else
      aR = *(const uint4*)(atile + (size_t)arow * K + k0 + acol);
#endif
      const float4* bp = (const float4*)(bbase + (size_t)(k0 + brow) * ldb);
#pragma unroll
      for (int t = 0; t < 4; ++t) bR[t] = bp[t];
      if (i + 2 < niter)
        __builtin_prefetch(bbase + (size_t)(((i + 2) << 5) + brow) * ldb, 0, 1);
    }

    Frag af[4], bfr[2];
#pragma unroll
    for (int mi = 0; mi < 4; ++mi) {
      const __bf16* ap = &lds_a[buf][wm * 64 + mi * 16 + lr][0];
#pragma unroll
      for (int j = 0; j < 8; ++j)
        af[mi].u[j] = *(const uint32_t*)(ap + aoff(j, kh));
    }
#pragma unroll
    for (int ni = 0; ni < 2; ++ni) {
      const __bf16* bp2 = &lds_b[wn * 32 + ni * 16 + lr][0];
#pragma unroll
      for (int j = 0; j < 8; ++j)
        bfr[ni].u[j] = *(const uint32_t*)(bp2 + boff(j, kh));
    }
#pragma unroll
    for (int mi = 0; mi < 4; ++mi)
#pragma unroll
      for (int ni = 0; ni < 2; ++ni)
        acc[mi][ni] = __builtin_amdgcn_wmma_f32_16x16x32_bf16(
            false, af[mi].v, false, bfr[ni].v, (short)0, acc[mi][ni],
            false, false);

    if (++i == niter) break;
    __syncthreads();
  }

#pragma unroll
  for (int mi = 0; mi < 4; ++mi)
#pragma unroll
    for (int ni = 0; ni < 2; ++ni) {
      int rbase = bm + wm * 64 + mi * 16 + kh * 8;
      int c = bn + wn * 32 + ni * 16 + lr;
      float* cp = C + (size_t)rbase * ldc + c;
      if (ADD) {
        float old[8];
#pragma unroll
        for (int v = 0; v < 8; ++v) old[v] = cp[(size_t)v * ldc];
#pragma unroll
        for (int v = 0; v < 8; ++v)
          cp[(size_t)v * ldc] = old[v] + acc[mi][ni][v];
      } else {
#pragma unroll
        for (int v = 0; v < 8; ++v) cp[(size_t)v * ldc] = acc[mi][ni][v];
      }
    }
}

// ------------------------------------------------------------------
// Fused gated-MLP GEMM: h[r][c] = bf16( up * silu(gate) )
// ------------------------------------------------------------------
__global__ __launch_bounds__(256) void gemm_mlp_kernel(
    const __bf16* __restrict__ A, const float* __restrict__ B,
    __bf16* __restrict__ Hout) {
  __shared__ __bf16 lds_a[128][32];
  __shared__ __bf16 lds_g[128][32];
  __shared__ __bf16 lds_u[128][32];
  const int tid  = threadIdx.x;
  const int lane = tid & 31;
  const int wave = tid >> 5;
  const int wm = wave >> 2;
  const int wn = wave & 3;
  const int lr = lane & 15;
  const int kh = lane >> 4;
  const int bm = blockIdx.y * 128;
  const int bn = blockIdx.x * 128;
  const int ldb = 2 * HID_;

  v8f zero8 = {0, 0, 0, 0, 0, 0, 0, 0};
  v8f accg[4][2], accu[4][2];
#pragma unroll
  for (int mi = 0; mi < 4; ++mi)
#pragma unroll
    for (int ni = 0; ni < 2; ++ni) { accg[mi][ni] = zero8; accu[mi][ni] = zero8; }

  const int arow = tid >> 1;
  const int acol = (tid & 1) * 16;
  const int brow = tid >> 3;
  const int bcol = (tid & 7) * 16;

  for (int k0 = 0; k0 < E_; k0 += 32) {
    *(uint4*)&lds_a[arow][acol] =
        *(const uint4*)(A + (size_t)(bm + arow) * E_ + k0 + acol);
    const float* bg = B + (size_t)(k0 + brow) * ldb + bn + bcol;
#pragma unroll
    for (int t = 0; t < 4; ++t) {
      float4 fg = ((const float4*)bg)[t];
      float4 fu = ((const float4*)(bg + HID_))[t];
      lds_g[bcol + 4 * t + 0][brow] = (__bf16)fg.x;
      lds_g[bcol + 4 * t + 1][brow] = (__bf16)fg.y;
      lds_g[bcol + 4 * t + 2][brow] = (__bf16)fg.z;
      lds_g[bcol + 4 * t + 3][brow] = (__bf16)fg.w;
      lds_u[bcol + 4 * t + 0][brow] = (__bf16)fu.x;
      lds_u[bcol + 4 * t + 1][brow] = (__bf16)fu.y;
      lds_u[bcol + 4 * t + 2][brow] = (__bf16)fu.z;
      lds_u[bcol + 4 * t + 3][brow] = (__bf16)fu.w;
    }
    __syncthreads();

    Frag af[4], bfr[2];
#pragma unroll
    for (int mi = 0; mi < 4; ++mi) {
      const __bf16* ap = &lds_a[wm * 64 + mi * 16 + lr][0];
#pragma unroll
      for (int j = 0; j < 8; ++j)
        af[mi].u[j] = *(const uint32_t*)(ap + aoff(j, kh));
    }
#pragma unroll
    for (int ni = 0; ni < 2; ++ni) {
      const __bf16* bp2 = &lds_g[wn * 32 + ni * 16 + lr][0];
#pragma unroll
      for (int j = 0; j < 8; ++j)
        bfr[ni].u[j] = *(const uint32_t*)(bp2 + boff(j, kh));
    }
#pragma unroll
    for (int mi = 0; mi < 4; ++mi)
#pragma unroll
      for (int ni = 0; ni < 2; ++ni)
        accg[mi][ni] = __builtin_amdgcn_wmma_f32_16x16x32_bf16(
            false, af[mi].v, false, bfr[ni].v, (short)0, accg[mi][ni],
            false, false);
#pragma unroll
    for (int ni = 0; ni < 2; ++ni) {
      const __bf16* bp2 = &lds_u[wn * 32 + ni * 16 + lr][0];
#pragma unroll
      for (int j = 0; j < 8; ++j)
        bfr[ni].u[j] = *(const uint32_t*)(bp2 + boff(j, kh));
    }
#pragma unroll
    for (int mi = 0; mi < 4; ++mi)
#pragma unroll
      for (int ni = 0; ni < 2; ++ni)
        accu[mi][ni] = __builtin_amdgcn_wmma_f32_16x16x32_bf16(
            false, af[mi].v, false, bfr[ni].v, (short)0, accu[mi][ni],
            false, false);
    __syncthreads();
  }

#pragma unroll
  for (int mi = 0; mi < 4; ++mi)
#pragma unroll
    for (int ni = 0; ni < 2; ++ni)
#pragma unroll
      for (int v = 0; v < 8; ++v) {
        int r = bm + wm * 64 + mi * 16 + v + kh * 8;
        int c = bn + wn * 32 + ni * 16 + lr;
        float g = accg[mi][ni][v];
        float u = accu[mi][ni][v];
        float sig = __builtin_amdgcn_rcpf(1.0f + __expf(-g));
        Hout[(size_t)r * HID_ + c] = (__bf16)(u * g * sig);
      }
}

// ------------------------------------------------------------------
__global__ __launch_bounds__(256) void rope_pack_kernel(
    const float* __restrict__ q, const float* __restrict__ k,
    const float* __restrict__ v, __bf16* __restrict__ qbf,
    __bf16* __restrict__ kbf, __bf16* __restrict__ vt) {
  int s = blockIdx.x;
  const float ln1e4 = 9.210340371976184f;
  for (int i = threadIdx.x; i < QDIM; i += 256) {
    int d = i & 127;
    int off = d & 63;
    float freq = __expf(-(float)off * (2.0f / 128.0f) * ln1e4);
    float ang = (float)s * freq;
    float c = __cosf(ang), sn = __sinf(ang);
    float sgn = (d < 64) ? -sn : sn;
    int base = i & ~127;
    float f  = q[(size_t)s * QDIM + i];
    float fs = q[(size_t)s * QDIM + base + ((d + 64) & 127)];
    qbf[(size_t)s * QDIM + i] = (__bf16)(c * f + sgn * fs);
  }
  for (int i = threadIdx.x; i < KDIM; i += 256) {
    int d = i & 127;
    int off = d & 63;
    float freq = __expf(-(float)off * (2.0f / 128.0f) * ln1e4);
    float ang = (float)s * freq;
    float c = __cosf(ang), sn = __sinf(ang);
    float sgn = (d < 64) ? -sn : sn;
    int base = i & ~127;
    float f  = k[(size_t)s * KDIM + i];
    float fs = k[(size_t)s * KDIM + base + ((d + 64) & 127)];
    kbf[(size_t)s * KDIM + i] = (__bf16)(c * f + sgn * fs);
  }
  for (int i = threadIdx.x; i < VDIM; i += 256)
    vt[(size_t)i * S_ + s] = (__bf16)v[(size_t)s * VDIM + i];
}

// ------------------------------------------------------------------
__global__ __launch_bounds__(32) void attn_kernel(
    const __bf16* __restrict__ qbf, const __bf16* __restrict__ kbf,
    const __bf16* __restrict__ vt, __bf16* __restrict__ abf) {
  const int qb = blockIdx.x;
  const int h  = blockIdx.y;
  const int kvh = h >> 2;
  const int lane = threadIdx.x & 31;
  const int lr = lane & 15;
  const int kh = lane >> 4;
  const int qbase = qb * 16;
  const float inv_sqrt_dk = 0.08838834764831845f;

  Frag qf[4];
  {
    const __bf16* qp = qbf + (size_t)(qbase + lr) * QDIM + h * DK_;
#pragma unroll
    for (int dc = 0; dc < 4; ++dc)
#pragma unroll
      for (int j = 0; j < 8; ++j)
        qf[dc].u[j] = *(const uint32_t*)(qp + dc * 32 + aoff(j, kh));
  }

  v8f zero8 = {0, 0, 0, 0, 0, 0, 0, 0};
  v8f o[8];
#pragma unroll
  for (int t = 0; t < 8; ++t) o[t] = zero8;
  float m[8], l[8];
#pragma unroll
  for (int v = 0; v < 8; ++v) { m[v] = -3.0e38f; l[v] = 0.0f; }

  __shared__ __bf16 lds_p[16][32];

  int jlo = qbase - (WIN_ - 1);
  if (jlo < 0) jlo = 0;
  jlo &= ~31;

  for (int j0 = jlo; j0 < qbase + 16; j0 += 32) {
    v8f sc[2];
#pragma unroll
    for (int t = 0; t < 2; ++t) {
      v8f s = zero8;
      const __bf16* kp = kbf + (size_t)(j0 + t * 16 + lr) * KDIM + kvh * DK_;
#pragma unroll
      for (int dc = 0; dc < 4; ++dc) {
        Frag kb;
#pragma unroll
        for (int j = 0; j < 8; ++j)
          kb.u[j] = *(const uint32_t*)(kp + dc * 32 + boff(j, kh));
        s = __builtin_amdgcn_wmma_f32_16x16x32_bf16(
            false, qf[dc].v, false, kb.v, (short)0, s, false, false);
      }
      sc[t] = s;
    }
    float pm[8];
#pragma unroll
    for (int v = 0; v < 8; ++v) {
      int i = qbase + v + kh * 8;
      int ja = j0 + lr, jb = j0 + 16 + lr;
      float a = sc[0][v] * inv_sqrt_dk;
      float b = sc[1][v] * inv_sqrt_dk;
      if (ja > i || i - ja >= WIN_) a = -1.0e9f;
      if (jb > i || i - jb >= WIN_) b = -1.0e9f;
      sc[0][v] = a; sc[1][v] = b;
      pm[v] = fmaxf(a, b);
    }
#pragma unroll
    for (int st = 1; st < 16; st <<= 1)
#pragma unroll
      for (int v = 0; v < 8; ++v)
        pm[v] = fmaxf(pm[v], __shfl_xor(pm[v], st, 32));
    float alpha[8], rs[8];
#pragma unroll
    for (int v = 0; v < 8; ++v) {
      float nm = fmaxf(m[v], pm[v]);
      alpha[v] = __expf(m[v] - nm);
      m[v] = nm;
      float p0 = __expf(sc[0][v] - nm);
      float p1 = __expf(sc[1][v] - nm);
      sc[0][v] = p0; sc[1][v] = p1;
      rs[v] = p0 + p1;
    }
#pragma unroll
    for (int st = 1; st < 16; st <<= 1)
#pragma unroll
      for (int v = 0; v < 8; ++v) rs[v] += __shfl_xor(rs[v], st, 32);
#pragma unroll
    for (int v = 0; v < 8; ++v) l[v] = l[v] * alpha[v] + rs[v];
#pragma unroll
    for (int ct = 0; ct < 8; ++ct)
#pragma unroll
      for (int v = 0; v < 8; ++v) o[ct][v] *= alpha[v];
    __syncthreads();
#pragma unroll
    for (int v = 0; v < 8; ++v) {
      lds_p[v + kh * 8][lr]      = (__bf16)sc[0][v];
      lds_p[v + kh * 8][16 + lr] = (__bf16)sc[1][v];
    }
    __syncthreads();
    Frag pa;
#pragma unroll
    for (int j = 0; j < 8; ++j)
      pa.u[j] = *(const uint32_t*)(&lds_p[lr][aoff(j, kh)]);
#pragma unroll
    for (int ct = 0; ct < 8; ++ct) {
      Frag vb;
      const __bf16* vp = vt + (size_t)(kvh * DK_ + ct * 16 + lr) * S_ + j0;
#pragma unroll
      for (int j = 0; j < 8; ++j)
        vb.u[j] = *(const uint32_t*)(vp + boff(j, kh));
      o[ct] = __builtin_amdgcn_wmma_f32_16x16x32_bf16(
          false, pa.v, false, vb.v, (short)0, o[ct], false, false);
    }
  }
#pragma unroll
  for (int ct = 0; ct < 8; ++ct)
#pragma unroll
    for (int v = 0; v < 8; ++v) {
      int row = qbase + v + kh * 8;
      int col = ct * 16 + lr;
      float inv_l = __builtin_amdgcn_rcpf(l[v]);
      abf[(size_t)row * ODIM + h * DH_ + col] = (__bf16)(o[ct][v] * inv_l);
    }
}

// ------------------------------------------------------------------
extern "C" void kernel_launch(void* const* d_in, const int* in_sizes, int n_in,
                              void* d_out, int out_size, void* d_ws,
                              size_t ws_size, hipStream_t stream) {
  (void)in_sizes; (void)n_in; (void)out_size; (void)ws_size;
  const int*   tokens = (const int*)  d_in[0];
  const float* table  = (const float*)d_in[1];
  const float* wq     = (const float*)d_in[2];
  const float* wk     = (const float*)d_in[3];
  const float* wv     = (const float*)d_in[4];
  const float* wo     = (const float*)d_in[5];
  const float* w_up   = (const float*)d_in[6];
  const float* w_down = (const float*)d_in[7];
  const float* w_voc  = (const float*)d_in[8];
  float* out = (float*)d_out;

  char* base = (char*)d_ws;
  size_t off = 0;
  auto alloc = [&](size_t bytes) -> void* {
    void* r = base + off;
    off += (bytes + 255) & ~(size_t)255;
    return r;
  };
  float*  x    = (float*) alloc((size_t)S_ * E_ * 4);
  __bf16* nbf  = (__bf16*)alloc((size_t)S_ * E_ * 2);
  float*  qf32 = (float*) alloc((size_t)S_ * QDIM * 4);
  float*  kf32 = (float*) alloc((size_t)S_ * KDIM * 4);
  float*  vf32 = (float*) alloc((size_t)S_ * VDIM * 4);
  __bf16* qbf  = (__bf16*)alloc((size_t)S_ * QDIM * 2);
  __bf16* kbf  = (__bf16*)alloc((size_t)S_ * KDIM * 2);
  __bf16* vtb  = (__bf16*)alloc((size_t)VDIM * S_ * 2);
  __bf16* abf  = (__bf16*)alloc((size_t)S_ * ODIM * 2);
  __bf16* hbf  = (__bf16*)alloc((size_t)S_ * HID_ * 2);

  dim3 b256(256);
  embed_kernel<<<dim3(S_), b256, 0, stream>>>(tokens, table, x);
  for (int l = 0; l < L_; ++l) {
    rmsnorm_kernel<<<dim3(S_), b256, 0, stream>>>(x, nbf);
    gemm_kernel<0><<<dim3(QDIM / 128, S_ / 128), b256, 0, stream>>>(
        nbf, wq + (size_t)l * E_ * QDIM, qf32, E_, QDIM, QDIM);
    gemm_kernel<0><<<dim3(KDIM / 128, S_ / 128), b256, 0, stream>>>(
        nbf, wk + (size_t)l * E_ * KDIM, kf32, E_, KDIM, KDIM);
    gemm_kernel<0><<<dim3(VDIM / 128, S_ / 128), b256, 0, stream>>>(
        nbf, wv + (size_t)l * E_ * VDIM, vf32, E_, VDIM, VDIM);
    rope_pack_kernel<<<dim3(S_), b256, 0, stream>>>(qf32, kf32, vf32, qbf,
                                                    kbf, vtb);
    attn_kernel<<<dim3(S_ / 16, H_), dim3(32), 0, stream>>>(qbf, kbf, vtb,
                                                            abf);
    gemm_kernel<1><<<dim3(E_ / 128, S_ / 128), b256, 0, stream>>>(
        abf, wo + (size_t)l * ODIM * E_, x, ODIM, E_, E_);
    rmsnorm_kernel<<<dim3(S_), b256, 0, stream>>>(x, nbf);
    gemm_mlp_kernel<<<dim3(HID_ / 128, S_ / 128), b256, 0, stream>>>(
        nbf, w_up + (size_t)l * E_ * 2 * HID_, hbf);
    gemm_kernel<1><<<dim3(E_ / 128, S_ / 128), b256, 0, stream>>>(
        hbf, w_down + (size_t)l * HID_ * E_, x, HID_, E_, E_);
  }
  rmsnorm_kernel<<<dim3(S_), b256, 0, stream>>>(x, nbf);
  gemm_kernel<0><<<dim3(VOC_ / 128, S_ / 128), b256, 0, stream>>>(
      nbf, w_voc, out, E_, VOC_, VOC_);
}
